// Linformer_72181220376956
// MI455X (gfx1250) — compile-verified, matrix operand order checked by reading, not run on hardware
//
#include <hip/hip_runtime.h>
#include <hip/hip_bf16.h>

// MI455X / gfx1250, wave32. Linformer pipeline collapsed algebraically
// (see analysis): colsum GEMVs + 4 GEMM instances of one WMMA kernel.
// v_wmma_f32_16x16x32_bf16: bf16 operands, f32 accumulate.
// Intermediates (Qs, attnS, MhT) stored bf16 to cut HBM traffic ~25%.

typedef __attribute__((ext_vector_type(16))) __bf16 v16bf;
typedef __attribute__((ext_vector_type(8)))  float  v8f;

#define TILE_M 128
#define TILE_N 128
#define TILE_K 32
#define LDS_STRIDE 34   // 32 data + 2 pad bf16 -> 68B row stride, conflict-free

// Gather one 16x32(bf16) WMMA A/B fragment for this lane from an LDS tile row.
// ISA 16-bit A-layout (wave32): VGPR v holds K pair at
//   k = (v>>2)*16 + (lane>>4)*8 + (v&3)*2   (B uses same packing, lane = column)
__device__ __forceinline__ v16bf frag_from_lds(const __bf16* rowBase, int hi) {
  union { v16bf v; unsigned int u[8]; } f;
#pragma unroll
  for (int vv = 0; vv < 8; ++vv) {
    int k = ((vv >> 2) << 4) | (hi << 3) | ((vv & 3) << 1);   // even -> 4B aligned
    f.u[vv] = *(const unsigned int*)(rowBase + k);
  }
  return f.v;
}

// Generic C[m,n] = sum_k A[m, aOff+k] * B[bRow0+n, k].
//   ABF/BBF : source element type (1 = bf16 straight copy, 0 = fp32 -> bf16 cvt)
//   CBF     : output element type (1 = bf16, 0 = fp32)
//   aColPerTileN : A's K-window starts at column n0 (block-diagonal stage)
//   bOffPerTileM : B row block tracks the M tile (per-head M_h build)
//   svec         : if non-null, C *= svec[(gm>>10)*1024 + gn] * escale
template<int ABF, int BBF, int CBF>
__global__ __launch_bounds__(256)
void gemm_wmma(const void* __restrict__ Ap, int lda,
               const void* __restrict__ Bp, int ldb,
               void* __restrict__ Cp, int ldc,
               int K, int aColPerTileN, int bOffPerTileM,
               const float* __restrict__ svec, float escale)
{
  __shared__ __bf16 As[TILE_M][LDS_STRIDE];
  __shared__ __bf16 Bs[TILE_N][LDS_STRIDE];

  const int m0   = blockIdx.x * TILE_M;
  const int n0   = blockIdx.y * TILE_N;
  const int tid  = threadIdx.x;
  const int lane = tid & 31;
  const int wid  = tid >> 5;
  const int waveM = (wid & 3) * 32;   // 4 waves down M
  const int waveN = (wid >> 2) * 64;  // 2 waves across N

  const int aOff  = aColPerTileN ? n0 : 0;
  const int bRow0 = bOffPerTileM ? m0 : n0;

  // 256 threads cover a 128x32 tile: 2 threads/row, 16 elements each
  const int tr = tid >> 1;
  const int tc = (tid & 1) * 16;
  const size_t aRowOff = (size_t)(m0 + tr)    * lda + aOff + tc;
  const size_t bRowOff = (size_t)(bRow0 + tr) * ldb + tc;

  v8f acc[2][4];
  v8f zero = {};
#pragma unroll
  for (int i = 0; i < 2; ++i)
#pragma unroll
    for (int j = 0; j < 4; ++j) acc[i][j] = zero;

  const int hi = lane >> 4;
  const int lr = lane & 15;

  for (int kt = 0; kt < K; kt += TILE_K) {
    if constexpr (ABF) {
      const __bf16* s = (const __bf16*)Ap + aRowOff + kt;   // 32B straight copy
#pragma unroll
      for (int i = 0; i < 8; ++i)
        *(unsigned int*)(&As[tr][tc + 2 * i]) = *(const unsigned int*)(s + 2 * i);
    } else {
      const float* s = (const float*)Ap + aRowOff + kt;
#pragma unroll
      for (int i = 0; i < 4; ++i) {
        float4 f = *(const float4*)(s + 4 * i);
        As[tr][tc + 4*i + 0] = (__bf16)f.x; As[tr][tc + 4*i + 1] = (__bf16)f.y;
        As[tr][tc + 4*i + 2] = (__bf16)f.z; As[tr][tc + 4*i + 3] = (__bf16)f.w;
      }
    }
    if constexpr (BBF) {
      const __bf16* s = (const __bf16*)Bp + bRowOff + kt;
#pragma unroll
      for (int i = 0; i < 8; ++i)
        *(unsigned int*)(&Bs[tr][tc + 2 * i]) = *(const unsigned int*)(s + 2 * i);
    } else {
      const float* s = (const float*)Bp + bRowOff + kt;
#pragma unroll
      for (int i = 0; i < 4; ++i) {
        float4 f = *(const float4*)(s + 4 * i);
        Bs[tr][tc + 4*i + 0] = (__bf16)f.x; Bs[tr][tc + 4*i + 1] = (__bf16)f.y;
        Bs[tr][tc + 4*i + 2] = (__bf16)f.z; Bs[tr][tc + 4*i + 3] = (__bf16)f.w;
      }
    }
    __syncthreads();

    v16bf af[2], bfv[4];
#pragma unroll
    for (int tm = 0; tm < 2; ++tm)
      af[tm] = frag_from_lds(&As[waveM + 16 * tm + lr][0], hi);
#pragma unroll
    for (int tn = 0; tn < 4; ++tn)
      bfv[tn] = frag_from_lds(&Bs[waveN + 16 * tn + lr][0], hi);

#pragma unroll
    for (int tm = 0; tm < 2; ++tm)
#pragma unroll
      for (int tn = 0; tn < 4; ++tn)
        acc[tm][tn] = __builtin_amdgcn_wmma_f32_16x16x32_bf16(
            false, af[tm], false, bfv[tn], (short)0, acc[tm][tn], false, false);

    __syncthreads();
  }

  // Epilogue: C/D layout -> VGPR r holds (M = r + 8*hi, N = lane&15)
#pragma unroll
  for (int tm = 0; tm < 2; ++tm) {
#pragma unroll
    for (int tn = 0; tn < 4; ++tn) {
      const int gn = n0 + waveN + 16 * tn + lr;
#pragma unroll
      for (int r = 0; r < 8; ++r) {
        const int gm = m0 + waveM + 16 * tm + (hi << 3) + r;
        float v = acc[tm][tn][r];
        if (svec) v *= svec[((gm >> 10) << 10) + gn] * escale;
        const size_t idx = (size_t)gm * ldc + gn;
        if constexpr (CBF) ((__bf16*)Cp)[idx] = (__bf16)v;
        else               ((float*)Cp)[idx]  = v;
      }
    }
  }
}

// wkcol[e] = sum_d Wk[d,e] ; wvcol[e] = sum_d Wv[d,e]   (1024 threads total)
__global__ void colsum_kernel(const float* __restrict__ Wk,
                              const float* __restrict__ Wv,
                              float* __restrict__ wkcol, float* __restrict__ wvcol)
{
  int e = blockIdx.x * blockDim.x + threadIdx.x;
  float sk = 0.f, sv = 0.f;
  for (int d = 0; d < 1024; ++d) {
    sk += Wk[d * 1024 + e];
    sv += Wv[d * 1024 + e];
  }
  wkcol[e] = sk;
  wvcol[e] = sv;
}

// ksv[row] = x[row,:]·wkcol ; vsv[row] = x[row,:]·wvcol  (one wave32 per row)
__global__ void ksvs_kernel(const float* __restrict__ x,
                            const float* __restrict__ wkcol,
                            const float* __restrict__ wvcol,
                            float* __restrict__ ksv, float* __restrict__ vsv)
{
  int row  = blockIdx.x * 8 + (threadIdx.x >> 5);
  int lane = threadIdx.x & 31;
  const float* xr = x + (size_t)row * 1024;
  float sk = 0.f, sv = 0.f;
  for (int e = lane; e < 1024; e += 32) {
    float xv = xr[e];
    sk += xv * wkcol[e];
    sv += xv * wvcol[e];
  }
#pragma unroll
  for (int off = 16; off > 0; off >>= 1) {
    sk += __shfl_down(sk, off, 32);
    sv += __shfl_down(sv, off, 32);
  }
  if (lane == 0) { ksv[row] = sk; vsv[row] = sv; }
}

extern "C" void kernel_launch(void* const* d_in, const int* in_sizes, int n_in,
                              void* d_out, int out_size, void* d_ws, size_t ws_size,
                              hipStream_t stream)
{
  (void)in_sizes; (void)n_in; (void)out_size; (void)ws_size;

  const float* x      = (const float*)d_in[0];   // [8,1024,1024] -> [8192,1024]
  const float* Wq     = (const float*)d_in[1];   // [1024,1024] (row = out ch)
  const float* Wk     = (const float*)d_in[2];
  const float* Wv     = (const float*)d_in[3];
  const float* Wo     = (const float*)d_in[4];
  const float* proj_k = (const float*)d_in[5];   // [1024,1024]
  const float* proj_v = (const float*)d_in[6];
  float* out = (float*)d_out;                    // [8192,1024] fp32

  // Workspace (byte offsets, all 16B aligned); total ~32.3 MB
  char* ws = (char*)d_ws;
  float*  ksv   = (float*)(ws + 0);              // 8192 f32
  float*  vsv   = (float*)(ws + 32768);          // 8192 f32
  float*  wkcol = (float*)(ws + 65536);          // 1024 f32
  float*  wvcol = (float*)(ws + 69632);          // 1024 f32
  __bf16* MhT   = (__bf16*)(ws + 73728);         // 1024x128 bf16 (row=h*128+dd, col=d)
  __bf16* Qs    = (__bf16*)(ws + 335872);        // 8192x1024 bf16
  __bf16* attnS = (__bf16*)(ws + 17113088);      // 8192x1024 bf16

  const float SCALE2 = 1.0f / 128.0f;            // (HEAD_D^-0.5)^2

  // 1) column sums of Wk/Wv (the einsum collapses k/v GEMMs to these)
  colsum_kernel<<<dim3(4), dim3(256), 0, stream>>>(Wk, Wv, wkcol, wvcol);
  // 2) per-(batch,position) scalars ksv/vsv
  ksvs_kernel<<<dim3(1024), dim3(256), 0, stream>>>(x, wkcol, wvcol, ksv, vsv);
  // 3) MhT[h*128+dd][d] = sum_kk proj_v[h*128+dd,kk] * proj_k[h*128+d,kk]
  //    (batch-independent; 8 head blocks via bOffPerTileM)
  gemm_wmma<0,0,1><<<dim3(8, 1), dim3(256), 0, stream>>>(
      proj_v, 1024, proj_k, 1024, MhT, 128, 1024, 0, 1, nullptr, 1.0f);
  // 4) Qs = (x @ Wq^T) * ksv[b, col] * SCALE^2  -> bf16
  gemm_wmma<0,0,1><<<dim3(64, 8), dim3(256), 0, stream>>>(
      x, 1024, Wq, 1024, Qs, 1024, 1024, 0, 0, ksv, SCALE2);
  // 5) attnS[:, h*128+dd] = (Qs[:, h*128 : h*128+128] @ MhT_h^T) * vsv[b, col]
  //    block-diagonal: K = 128, A column window tracks the N tile
  gemm_wmma<1,1,1><<<dim3(64, 8), dim3(256), 0, stream>>>(
      Qs, 1024, MhT, 128, attnS, 1024, 128, 1, 0, vsv, 1.0f);
  // 6) out = attnS @ Wo^T  -> fp32
  gemm_wmma<1,0,0><<<dim3(64, 8), dim3(256), 0, stream>>>(
      attnS, 1024, Wo, 1024, out, 1024, 1024, 0, 0, nullptr, 1.0f);
}
